// GCN_27986006901493
// MI455X (gfx1250) — matrix operand order repaired and checked by reading, not compiled, not run on hardware
//
#include <hip/hip_runtime.h>

// ---------------------------------------------------------------------------
// GCN on MI455X (gfx1250, wave32).
// Roofline: node tables (10000x128 f32 = 5.1MB) are L2-resident (192MB L2),
// so we fuse the GEMM epilogue -> next layer's scatter-add and never write
// the 164MB edge activations to HBM. HBM traffic ~= 164MB in (non-temporal
// streamed) + 82MB out (non-temporal stores). Everything else is L2 atomics.
// Matrix ops: v_wmma_f32_16x16x32_f16 (f16 in, fp32 accumulate, K=128=4x32).
// ---------------------------------------------------------------------------

#define N_NODES 10000
#define N_EDGES 320000
#define F       128

typedef __attribute__((ext_vector_type(16))) _Float16 v16h;
typedef __attribute__((ext_vector_type(8)))  _Float16 v8h;
typedef __attribute__((ext_vector_type(8)))  float    v8f;
typedef __attribute__((ext_vector_type(4)))  float    v4f;

__global__ void zero_f(float* __restrict__ p, int n) {
    int t = blockIdx.x * blockDim.x + threadIdx.x;
    if (t < n) p[t] = 0.f;
}

// acc[dst[e]] += edge_feats[e]  (the only HBM-heavy read: stream it
// non-temporally so it doesn't evict the L2-resident node tables).
// Lane with q==0 also bumps the in-degree counter (fused deg pass).
__global__ void scatter_edge_kernel(const float* __restrict__ ef,
                                    const int* __restrict__ dst,
                                    float* __restrict__ acc,
                                    float* __restrict__ deg) {
    int t = blockIdx.x * blockDim.x + threadIdx.x;   // N_EDGES * 32 threads
    int e = t >> 5, q = t & 31;
    if (e >= N_EDGES) return;
    int d = dst[e];
    if (q == 0) atomicAdd(&deg[d], 1.0f);
    const v4f v = __builtin_nontemporal_load((const v4f*)(ef + (size_t)e * F + q * 4));
    float* o = acc + (size_t)d * F + q * 4;
    atomicAdd(o + 0, v.x); atomicAdd(o + 1, v.y);
    atomicAdd(o + 2, v.z); atomicAdd(o + 3, v.w);
}

// node_a = acc / deg ; acc = 0 ; node_b = 0   (all L2-resident)
__global__ void mean_zero_kernel(float* __restrict__ acc,
                                 const float* __restrict__ deg,
                                 float* __restrict__ na,
                                 float* __restrict__ nb) {
    int t = blockIdx.x * blockDim.x + threadIdx.x;
    if (t < N_NODES * F) {
        na[t] = acc[t] / deg[t >> 7];   // F == 128
        acc[t] = 0.f;
        nb[t]  = 0.f;
    }
}

// node_b[dst[e]] += node_a[src[e]]   (L2 atomics, 5.1MB tables)
__global__ void prop_kernel(const float* __restrict__ na,
                            const int* __restrict__ src,
                            const int* __restrict__ dst,
                            float* __restrict__ nb) {
    int t = blockIdx.x * blockDim.x + threadIdx.x;   // N_EDGES * 32 threads
    int e = t >> 5, q = t & 31;
    if (e >= N_EDGES) return;
    int s = src[e], d = dst[e];
    const float4 v = *(const float4*)(na + (size_t)s * F + q * 4);
    float* o = nb + (size_t)d * F + q * 4;
    atomicAdd(o + 0, v.x); atomicAdd(o + 1, v.y);
    atomicAdd(o + 2, v.z); atomicAdd(o + 3, v.w);
}

__global__ void convert_w_kernel(const float* __restrict__ W,
                                 _Float16* __restrict__ Wh, int n) {
    int t = blockIdx.x * blockDim.x + threadIdx.x;
    if (t < n) Wh[t] = (_Float16)W[t];
}

// Fused: gather edge tile 0.5*(nb[src]+nb[dst]) -> f16 LDS -> WMMA chain
// (K = 128 = 4 x 32) -> +bias [-> ReLU] -> either atomic scatter-add into
// next layer's node accumulator (L2), or non-temporal store to d_out.
// Block = 128 threads = 4 waves; 16 edge rows per block; wave covers
// N-tiles nt = wave, wave+4, ...
template <bool RELU, bool SCATTER>
__global__ __launch_bounds__(128)
void gemm_fused(const float* __restrict__ nb,
                const int* __restrict__ src,
                const int* __restrict__ dst,
                const _Float16* __restrict__ Wh,   // [nout][128] row-major f16
                const float* __restrict__ bias,
                int NT,                            // nout / 16
                float* __restrict__ out) {
    __shared__ __align__(16) _Float16 As[16][136];  // stride 136 halves: conflict-free
    __shared__ int sdst[16];

    const int tid   = threadIdx.x;
    const int ebase = blockIdx.x * 16;

    // ---- stage A tile: 16 rows x 128 cols, 8 threads per row ----
    {
        const int r = tid >> 3, chunk = tid & 7, c0 = chunk * 16;
        const int e = ebase + r;
        const int s = src[e], d = dst[e];
        if (chunk == 0) sdst[r] = d;
        const float4* ps = (const float4*)(nb + (size_t)s * F + c0);
        const float4* pd = (const float4*)(nb + (size_t)d * F + c0);
#pragma unroll
        for (int j = 0; j < 4; ++j) {
            float4 a = ps[j], b = pd[j];
            As[r][c0 + 4 * j + 0] = (_Float16)(0.5f * (a.x + b.x));
            As[r][c0 + 4 * j + 1] = (_Float16)(0.5f * (a.y + b.y));
            As[r][c0 + 4 * j + 2] = (_Float16)(0.5f * (a.z + b.z));
            As[r][c0 + 4 * j + 3] = (_Float16)(0.5f * (a.w + b.w));
        }
    }
    __syncthreads();

    const int lane = tid & 31;
    const int wave = tid >> 5;
    const int n    = lane & 15;     // also A row index m for fragment load
    const int hi   = lane >> 4;
    const int nout = NT * 16;

    for (int nt = wave; nt < NT; nt += 4) {
        v8f c = {};
        const _Float16* wrow = Wh + (size_t)(nt * 16 + n) * F;
#pragma unroll
        for (int kt = 0; kt < 4; ++kt) {
            // A fragment (16x32 f16): lane m(=n), idx i -> k = 32kt + 16*hi + i (contiguous)
            const v8h a0 = *(const v8h*)&As[n][kt * 32 + hi * 16];
            const v8h a1 = *(const v8h*)&As[n][kt * 32 + hi * 16 + 8];
            v16h a;
#pragma unroll
            for (int i = 0; i < 8; ++i) { a[i] = a0[i]; a[i + 8] = a1[i]; }
            // B fragment (32x16 f16): lane col n, idx i -> k = 32kt + 16*hi + i (contiguous)
            const v16h b = *(const v16h*)&wrow[kt * 32 + hi * 16];
            c = __builtin_amdgcn_wmma_f32_16x16x32_f16(
                    false, a, false, b, (short)0, c, false, false);
        }
        const float bv = bias[nt * 16 + n];
#pragma unroll
        for (int r = 0; r < 8; ++r) {
            float o = c[r] + bv;
            if (RELU) o = (o > 0.f) ? o : 0.f;
            const int m = r + 8 * hi;      // C/D layout: row = vgpr + 8*(lane>=16)
            if (SCATTER) {
                atomicAdd(&out[(size_t)sdst[m] * F + nt * 16 + n], o);
            } else {
                __builtin_nontemporal_store(
                    o, &out[(size_t)(ebase + m) * nout + nt * 16 + n]);
            }
        }
    }
}

extern "C" void kernel_launch(void* const* d_in, const int* in_sizes, int n_in,
                              void* d_out, int out_size, void* d_ws, size_t ws_size,
                              hipStream_t stream) {
    (void)in_sizes; (void)n_in; (void)out_size; (void)ws_size;

    const float* ef  = (const float*)d_in[0];
    const int*   src = (const int*)d_in[1];
    const int*   dst = (const int*)d_in[2];
    const float* W[5] = {(const float*)d_in[3], (const float*)d_in[5],
                         (const float*)d_in[7], (const float*)d_in[9],
                         (const float*)d_in[11]};
    const float* B[5] = {(const float*)d_in[4], (const float*)d_in[6],
                         (const float*)d_in[8], (const float*)d_in[10],
                         (const float*)d_in[12]};
    float* out = (float*)d_out;

    // workspace carve-out (~15.5 MB)
    char* ws = (char*)d_ws;
    auto carve = [&](size_t bytes) {
        void* p = (void*)ws;
        ws += (bytes + 255) & ~(size_t)255;
        return p;
    };
    float*    deg = (float*)carve((size_t)N_NODES * 4);
    float*    acc = (float*)carve((size_t)N_NODES * F * 4);
    float*    na  = (float*)carve((size_t)N_NODES * F * 4);
    float*    nbuf= (float*)carve((size_t)N_NODES * F * 4);
    _Float16* Wh  = (_Float16*)carve((size_t)F * F * 2);

    zero_f<<<(N_NODES + 255) / 256, 256, 0, stream>>>(deg, N_NODES);
    zero_f<<<(N_NODES * F + 255) / 256, 256, 0, stream>>>(acc, N_NODES * F);
    // layer-1 step 1: streamed scatter-add of edge features into acc (+deg)
    scatter_edge_kernel<<<(N_EDGES * 32) / 256, 256, 0, stream>>>(ef, dst, acc, deg);

    for (int l = 0; l < 5; ++l) {
        const int nout = (l == 4) ? 64 : F;
        mean_zero_kernel<<<(N_NODES * F + 255) / 256, 256, 0, stream>>>(acc, deg, na, nbuf);
        prop_kernel<<<(N_EDGES * 32) / 256, 256, 0, stream>>>(na, src, dst, nbuf);
        convert_w_kernel<<<(nout * F + 255) / 256, 256, 0, stream>>>(W[l], Wh, nout * F);
        if (l < 4) {
            // ReLU, and scatter output edges straight into next layer's acc
            gemm_fused<true, true><<<N_EDGES / 16, 128, 0, stream>>>(
                nbuf, src, dst, Wh, B[l], nout / 16, acc);
        } else {
            // last layer: no ReLU, write (N_EDGES x 64) logits to d_out
            gemm_fused<false, false><<<N_EDGES / 16, 128, 0, stream>>>(
                nbuf, src, dst, Wh, B[l], nout / 16, out);
        }
    }
}
